// NVFP4FakeQuantizedMoE_63402307224412
// MI455X (gfx1250) — matrix operand order repaired and checked by reading, not compiled, not run on hardware
//
#include <hip/hip_runtime.h>
#include <hip/hip_bf16.h>

typedef unsigned short u16;
typedef unsigned int   u32;

typedef __attribute__((ext_vector_type(16))) __bf16 bf16x16;
typedef __attribute__((ext_vector_type(8)))  float  f32x8;
typedef int v4i __attribute__((vector_size(16)));   // matches builtin's pointee type

union Frag { uint4 q[2]; bf16x16 v; };

// Problem constants (from setup_inputs)
constexpr int TT  = 2048;   // tokens
constexpr int EE  = 8;      // experts
constexpr int HH  = 1024;   // hidden
constexpr int IS  = 1024;   // intermediate
constexpr int KK  = 2;      // top-k
constexpr int CAP = 2 * (TT * KK / EE); // 1024

constexpr int LROW = 40;    // LDS row pitch in elements (32 data + 8 pad): 80B, 16B-aligned,
                            // 20-dword stride -> conflict-free 16-lane b128 reads

// Workspace layout (bytes)
constexpr size_t OFF_AMAX = 0;                       // u32[18]: [0]=x [1..8]=w1 [9..16]=w2 [17]=act
constexpr size_t OFF_POS  = 256;                     // int[T*K]
constexpr size_t OFF_XQ   = 64 * 1024;               // bf16 (T,H)
constexpr size_t XQ_B     = (size_t)TT * HH * 2;
constexpr size_t OFF_W1Q  = OFF_XQ + XQ_B;           // bf16 (E,2I,H)
constexpr size_t W1Q_B    = (size_t)EE * 2 * IS * HH * 2;
constexpr size_t OFF_W2Q  = OFF_W1Q + W1Q_B;         // bf16 (E,H,I)
constexpr size_t W2Q_B    = (size_t)EE * HH * IS * 2;
constexpr size_t OFF_XD   = OFF_W2Q + W2Q_B;         // bf16 (E,CAP,H) dispatched
constexpr size_t XD_B     = (size_t)EE * CAP * HH * 2;
constexpr size_t OFF_ACTB = OFF_XD + XD_B;           // bf16 (E,CAP,I) act (bf16-rounded)
constexpr size_t ACT_B    = (size_t)EE * CAP * IS * 2;
constexpr size_t OFF_ACTS = OFF_ACTB + ACT_B;        // bf16 (E,CAP,I) act quant-scaled
constexpr size_t OFF_H2   = OFF_ACTS + ACT_B;        // f32  (E,CAP,H)

// ---------------- numeric helpers ----------------

__device__ __forceinline__ float bf16r(float x) {   // RNE to bf16, back to f32
  u32 u = __float_as_uint(x);
  u += 0x7fffu + ((u >> 16) & 1u);
  return __uint_as_float(u & 0xffff0000u);
}
__device__ __forceinline__ u16 bf16b(float x) {
  u32 u = __float_as_uint(x);
  u += 0x7fffu + ((u >> 16) & 1u);
  return (u16)(u >> 16);
}
__device__ __forceinline__ float frombf16(u16 b) {
  return __uint_as_float(((u32)b) << 16);
}

// round non-negative float to FP8 E4M3 grid (RNE), saturating at 448
__device__ __forceinline__ float e4m3q(float v) {
  if (!(v > 0.f)) return 0.f;
  if (v >= 448.f) return 448.f;
  u32 bits = __float_as_uint(v);
  int ex = (int)((bits >> 23) & 0xff) - 127;
  if (ex < -6) ex = -6;                                    // denormal quantum 2^-9
  float step = __uint_as_float((u32)(ex - 3 + 127) << 23); // 2^(ex-3)
  float q = rintf(v / step) * step;
  if (q > 448.f) q = 448.f;
  return q;
}

// FP4 E2M1 quantization, matches searchsorted(side='left') semantics
__device__ __forceinline__ float e2m1q(float v) {
  float a = fabsf(v);
  float q;
  if      (a <= 0.25f) q = 0.0f;
  else if (a <= 0.75f) q = 0.5f;
  else if (a <= 1.25f) q = 1.0f;
  else if (a <= 1.75f) q = 1.5f;
  else if (a <= 2.5f)  q = 2.0f;
  else if (a <= 3.5f)  q = 3.0f;
  else if (a <= 5.0f)  q = 4.0f;
  else                 q = 6.0f;
  return copysignf(q, v);
}

__device__ __forceinline__ f32x8 wmma_bf16(const Frag& a, const Frag& b, f32x8 c) {
  return __builtin_amdgcn_wmma_f32_16x16x32_bf16(false, a.v, false, b.v, (short)0, c, false, false);
}

// ---- async global->LDS copy (16B per lane), ASYNCcnt tracked ----
__device__ __forceinline__ void async_cp16(const u16* g, u16* l) {
#if __has_builtin(__builtin_amdgcn_global_load_async_to_lds_b128)
  __builtin_amdgcn_global_load_async_to_lds_b128(
      (__attribute__((address_space(1))) v4i*)(v4i*)g,
      (__attribute__((address_space(3))) v4i*)(v4i*)l, 0, 0);
#else
  u32 laddr = (u32)(size_t)l;  // generic(shared) low 32 bits = LDS offset
  asm volatile("global_load_async_to_lds_b128 %0, %1, off"
               :: "v"(laddr), "v"(g) : "memory");
#endif
}
__device__ __forceinline__ void wait_async0() {
#if __has_builtin(__builtin_amdgcn_s_wait_asynccnt)
  __builtin_amdgcn_s_wait_asynccnt(0);
#else
  asm volatile("s_wait_asynccnt 0x0" ::: "memory");
#endif
}

// ---------------- stage kernels ----------------

// per-group amax of |bf16(x)| (nan -> 0); grid.y = group
__global__ void amax_f32_kernel(const float* __restrict__ src, u32* __restrict__ out, size_t epg) {
  src += (size_t)blockIdx.y * epg;
  float m = 0.f;
  for (size_t i = (size_t)blockIdx.x * blockDim.x + threadIdx.x; i < epg;
       i += (size_t)gridDim.x * blockDim.x) {
    float v = fabsf(bf16r(src[i]));
    if (!(v == v)) v = 0.f;
    m = fmaxf(m, v);
  }
  __shared__ float red[256];
  red[threadIdx.x] = m;
  __syncthreads();
  for (int s = 128; s > 0; s >>= 1) {
    if ((int)threadIdx.x < s) red[threadIdx.x] = fmaxf(red[threadIdx.x], red[threadIdx.x + s]);
    __syncthreads();
  }
  if (threadIdx.x == 0) atomicMax(out + blockIdx.y, __float_as_uint(red[0]));
}

// fp4 block quant of f32 src -> bf16 of (q * block_sf)  (1/gsf folded into GEMM epilogue)
__global__ void quant_f32_kernel(const float* __restrict__ src, u16* __restrict__ dst,
                                 const u32* __restrict__ amaxb, size_t epg, size_t nblk) {
  size_t b = (size_t)blockIdx.x * blockDim.x + threadIdx.x;
  if (b >= nblk) return;
  size_t base = b * 16;
  float amax = __uint_as_float(amaxb[base / epg]);
  float gsf = 2688.0f / amax;            // 448*6/amax
  float xs[16], bm = 0.f;
#pragma unroll
  for (int i = 0; i < 16; i++) {
    float v = bf16r(src[base + i]) * gsf;
    xs[i] = v;
    bm = fmaxf(bm, fabsf(v));
  }
  float bsf = e4m3q(bm / 6.0f);
  float den = fmaxf(bsf, 0.015625f);     // 2^-6 tiny clamp
#pragma unroll
  for (int i = 0; i < 16; i++) {
    float q = e2m1q(xs[i] / den);
    dst[base + i] = bf16b(q * bsf);      // exact in bf16 (<=5 mantissa bits)
  }
}

// same but bf16 source (for the activation)
__global__ void quant_bf16_kernel(const u16* __restrict__ src, u16* __restrict__ dst,
                                  const u32* __restrict__ amaxb, size_t epg, size_t nblk) {
  size_t b = (size_t)blockIdx.x * blockDim.x + threadIdx.x;
  if (b >= nblk) return;
  size_t base = b * 16;
  float amax = __uint_as_float(amaxb[base / epg]);
  float gsf = 2688.0f / amax;
  float xs[16], bm = 0.f;
#pragma unroll
  for (int i = 0; i < 16; i++) {
    float v = frombf16(src[base + i]) * gsf;
    xs[i] = v;
    bm = fmaxf(bm, fabsf(v));
  }
  float bsf = e4m3q(bm / 6.0f);
  float den = fmaxf(bsf, 0.015625f);
#pragma unroll
  for (int i = 0; i < 16; i++) {
    float q = e2m1q(xs[i] / den);
    dst[base + i] = bf16b(q * bsf);
  }
}

// stable per-expert rank of each (token,k) entry in flat order
__global__ void route_kernel(const int* __restrict__ ri, int* __restrict__ pos) {
  int e = threadIdx.x;
  if (e >= EE) return;
  int c = 0;
  for (int i = 0; i < TT * KK; i++)
    if (ri[i] == e) pos[i] = c++;
}

// copy quantized token rows into (E,CAP,H) dispatch buffer
__global__ void scatter_x_kernel(const u16* __restrict__ xq, const int* __restrict__ ri,
                                 const int* __restrict__ pos, u16* __restrict__ xd) {
  int i = blockIdx.x;
  int p = pos[i];
  if (p >= CAP) return;
  int e = ri[i];
  int t = i / KK;
  const uint2* s = (const uint2*)(xq + (size_t)t * HH);
  uint2* d = (uint2*)(xd + ((size_t)e * CAP + p) * HH);
  d[threadIdx.x] = s[threadIdx.x];  // 256 threads * 4 bf16 = 1024 = H
}

// -------- GEMM1: async-LDS staged, fused SiLU-gate, 32x32 per wave --------
// Block tile: M=128, N=64 value + 64 gate. LDS stage: A rows 0-127, B rows 128-255
// (128..191 = value cols, 192..255 = gate cols), row pitch LROW.

__global__ void __launch_bounds__(256, 1)
gemm1_act_kernel(const u16* __restrict__ X, const u16* __restrict__ W1,
                 u16* __restrict__ actB, const u32* __restrict__ amaxv,
                 u32* __restrict__ amax_act) {
  __shared__ __align__(16) u16 sm[2][256 * LROW];
  const int tid = threadIdx.x;
  const int lane = tid & 31, wv = tid >> 5;
  const int wm = wv & 3, wn = wv >> 2;
  const int e = blockIdx.z;
  const int nBlk = blockIdx.x * 64;
  const int mBlk = blockIdx.y * 128;
  const int r16 = lane & 15, hi = lane >> 4;

  const u16* Xe = X  + (size_t)e * CAP * HH;
  const u16* We = W1 + (size_t)e * (2 * IS) * HH;

  // staging addresses for this thread: 4 async 16B copies per stage
  const int srow = tid >> 2;            // 0..63
  const int sch  = (tid & 3) * 8;       // 0,8,16,24 elements
  const u16* gA0 = Xe + (size_t)(mBlk + srow) * HH + sch;
  const u16* gA1 = Xe + (size_t)(mBlk + 64 + srow) * HH + sch;
  const u16* gBV = We + (size_t)(nBlk + srow) * HH + sch;
  const u16* gBG = We + (size_t)(IS + nBlk + srow) * HH + sch;

  f32x8 aV[2][2] = {{{0,0,0,0,0,0,0,0},{0,0,0,0,0,0,0,0}},
                    {{0,0,0,0,0,0,0,0},{0,0,0,0,0,0,0,0}}};
  f32x8 aG[2][2] = {{{0,0,0,0,0,0,0,0},{0,0,0,0,0,0,0,0}},
                    {{0,0,0,0,0,0,0,0},{0,0,0,0,0,0,0,0}}};

  auto stage = [&](int s, int k0) {
    u16* b = &sm[s][0];
    async_cp16(gA0 + k0, b + (size_t)srow * LROW + sch);
    async_cp16(gA1 + k0, b + (size_t)(64 + srow) * LROW + sch);
    async_cp16(gBV + k0, b + (size_t)(128 + srow) * LROW + sch);
    async_cp16(gBG + k0, b + (size_t)(192 + srow) * LROW + sch);
  };

  auto consume = [&](int s) {
    const u16* b = &sm[s][0];
    Frag a0, a1, bv0, bv1, bg0, bg1;
    const u16* ap = b + (size_t)(wm * 32 + r16) * LROW + hi * 16;
    a0.q[0] = *(const uint4*)(ap);
    a0.q[1] = *(const uint4*)(ap + 8);
    const u16* ap1 = ap + 16 * LROW;
    a1.q[0] = *(const uint4*)(ap1);
    a1.q[1] = *(const uint4*)(ap1 + 8);
    const u16* vp = b + (size_t)(128 + wn * 32 + r16) * LROW + hi * 16;
    bv0.q[0] = *(const uint4*)(vp);
    bv0.q[1] = *(const uint4*)(vp + 8);
    const u16* vp1 = vp + 16 * LROW;
    bv1.q[0] = *(const uint4*)(vp1);
    bv1.q[1] = *(const uint4*)(vp1 + 8);
    const u16* gp = vp + 64 * LROW;
    bg0.q[0] = *(const uint4*)(gp);
    bg0.q[1] = *(const uint4*)(gp + 8);
    const u16* gp1 = gp + 16 * LROW;
    bg1.q[0] = *(const uint4*)(gp1);
    bg1.q[1] = *(const uint4*)(gp1 + 8);
    aV[0][0] = wmma_bf16(a0, bv0, aV[0][0]);
    aV[0][1] = wmma_bf16(a0, bv1, aV[0][1]);
    aV[1][0] = wmma_bf16(a1, bv0, aV[1][0]);
    aV[1][1] = wmma_bf16(a1, bv1, aV[1][1]);
    aG[0][0] = wmma_bf16(a0, bg0, aG[0][0]);
    aG[0][1] = wmma_bf16(a0, bg1, aG[0][1]);
    aG[1][0] = wmma_bf16(a1, bg0, aG[1][0]);
    aG[1][1] = wmma_bf16(a1, bg1, aG[1][1]);
  };

  stage(0, 0);
  wait_async0();
  __syncthreads();
#pragma unroll 2
  for (int k0 = 0; k0 < HH; k0 += 32) {
    const int cur = (k0 >> 5) & 1;
    if (k0 + 32 < HH) stage(cur ^ 1, k0 + 32);   // prefetch next chunk async
    consume(cur);                                 // ds reads + 8 WMMAs
    wait_async0();
    __syncthreads();
  }

  // epilogue: undo gsf factors, SiLU-gate, bf16 round, store, track amax
  const float ax = __uint_as_float(amaxv[0]);
  const float aw = __uint_as_float(amaxv[1 + e]);
  const float s1s = (ax * aw) * (1.0f / (2688.0f * 2688.0f));
  float lm = 0.f;
  u16* oe = actB + (size_t)e * CAP * IS;
#pragma unroll
  for (int mi = 0; mi < 2; mi++) {
#pragma unroll
    for (int ni = 0; ni < 2; ni++) {
#pragma unroll
      for (int r = 0; r < 8; r++) {
        const int row = mBlk + wm * 32 + mi * 16 + r + hi * 8;
        const int col = nBlk + wn * 32 + ni * 16 + r16;
        float v = aV[mi][ni][r] * s1s, g = aG[mi][ni][r] * s1s;
        float a = v * g / (1.0f + __expf(-g));
        u16 bb = bf16b(a);
        oe[(size_t)row * IS + col] = bb;
        lm = fmaxf(lm, fabsf(frombf16(bb)));
      }
    }
  }
  __shared__ float red[256];
  red[tid] = lm;
  __syncthreads();
  for (int s = 128; s > 0; s >>= 1) {
    if (tid < s) red[tid] = fmaxf(red[tid], red[tid + s]);
    __syncthreads();
  }
  if (tid == 0) atomicMax(amax_act, __float_as_uint(red[0]));
}

// -------- GEMM2: async-LDS staged, 32x32 per wave --------
// Block tile M=128, N=64. LDS stage: A rows 0-127, B rows 128-191.

__global__ void __launch_bounds__(256, 1)
gemm2_kernel(const u16* __restrict__ A, const u16* __restrict__ W2,
             float* __restrict__ h2, const u32* __restrict__ amaxv) {
  __shared__ __align__(16) u16 sm[2][192 * LROW];
  const int tid = threadIdx.x;
  const int lane = tid & 31, wv = tid >> 5;
  const int wm = wv & 3, wn = wv >> 2;
  const int e = blockIdx.z;
  const int nBlk = blockIdx.x * 64;
  const int mBlk = blockIdx.y * 128;
  const int r16 = lane & 15, hi = lane >> 4;

  const u16* Ae = A  + (size_t)e * CAP * IS;
  const u16* We = W2 + (size_t)e * HH * IS;

  const int srow = tid >> 2;
  const int sch  = (tid & 3) * 8;
  const u16* gA0 = Ae + (size_t)(mBlk + srow) * IS + sch;
  const u16* gA1 = Ae + (size_t)(mBlk + 64 + srow) * IS + sch;
  const u16* gB  = We + (size_t)(nBlk + srow) * IS + sch;

  f32x8 c[2][2] = {{{0,0,0,0,0,0,0,0},{0,0,0,0,0,0,0,0}},
                   {{0,0,0,0,0,0,0,0},{0,0,0,0,0,0,0,0}}};

  auto stage = [&](int s, int k0) {
    u16* b = &sm[s][0];
    async_cp16(gA0 + k0, b + (size_t)srow * LROW + sch);
    async_cp16(gA1 + k0, b + (size_t)(64 + srow) * LROW + sch);
    async_cp16(gB + k0,  b + (size_t)(128 + srow) * LROW + sch);
  };

  auto consume = [&](int s) {
    const u16* b = &sm[s][0];
    Frag a0, a1, b0, b1;
    const u16* ap = b + (size_t)(wm * 32 + r16) * LROW + hi * 16;
    a0.q[0] = *(const uint4*)(ap);
    a0.q[1] = *(const uint4*)(ap + 8);
    const u16* ap1 = ap + 16 * LROW;
    a1.q[0] = *(const uint4*)(ap1);
    a1.q[1] = *(const uint4*)(ap1 + 8);
    const u16* bp = b + (size_t)(128 + wn * 32 + r16) * LROW + hi * 16;
    b0.q[0] = *(const uint4*)(bp);
    b0.q[1] = *(const uint4*)(bp + 8);
    const u16* bp1 = bp + 16 * LROW;
    b1.q[0] = *(const uint4*)(bp1);
    b1.q[1] = *(const uint4*)(bp1 + 8);
    c[0][0] = wmma_bf16(a0, b0, c[0][0]);
    c[0][1] = wmma_bf16(a0, b1, c[0][1]);
    c[1][0] = wmma_bf16(a1, b0, c[1][0]);
    c[1][1] = wmma_bf16(a1, b1, c[1][1]);
  };

  stage(0, 0);
  wait_async0();
  __syncthreads();
#pragma unroll 2
  for (int k0 = 0; k0 < IS; k0 += 32) {
    const int cur = (k0 >> 5) & 1;
    if (k0 + 32 < IS) stage(cur ^ 1, k0 + 32);
    consume(cur);
    wait_async0();
    __syncthreads();
  }

  const float aa = __uint_as_float(amaxv[17]);
  const float aw = __uint_as_float(amaxv[9 + e]);
  const float s2 = (aa * aw) * (1.0f / (2688.0f * 2688.0f));
  float* oe = h2 + (size_t)e * CAP * HH;
#pragma unroll
  for (int mi = 0; mi < 2; mi++) {
#pragma unroll
    for (int ni = 0; ni < 2; ni++) {
#pragma unroll
      for (int r = 0; r < 8; r++) {
        const int row = mBlk + wm * 32 + mi * 16 + r + hi * 8;
        const int col = nBlk + wn * 32 + ni * 16 + r16;
        oe[(size_t)row * HH + col] = c[mi][ni][r] * s2;
      }
    }
  }
}

// out[t,h] = sum_k rw[t, e(t,k)] * h2[e(t,k), pos(t,k), h]
__global__ void gather_out_kernel(const float* __restrict__ h2, const int* __restrict__ ri,
                                  const int* __restrict__ pos, const float* __restrict__ rw,
                                  float* __restrict__ out) {
  size_t idx = (size_t)blockIdx.x * blockDim.x + threadIdx.x;
  if (idx >= (size_t)TT * HH) return;
  int t = (int)(idx / HH);
  int h = (int)(idx % HH);
  float acc = 0.f;
#pragma unroll
  for (int k = 0; k < KK; k++) {
    int fi = t * KK + k;
    int e = ri[fi];
    int p = pos[fi];
    if (p < CAP) {
      float w = rw[t * EE + e];
      acc += w * h2[(((size_t)e * CAP) + p) * HH + h];
    }
  }
  out[idx] = acc;
}

// ---------------- driver ----------------

extern "C" void kernel_launch(void* const* d_in, const int* in_sizes, int n_in,
                              void* d_out, int out_size, void* d_ws, size_t ws_size,
                              hipStream_t stream) {
  (void)in_sizes; (void)n_in; (void)out_size; (void)ws_size;
  const float* hs = (const float*)d_in[0];
  const float* rw = (const float*)d_in[1];
  const float* w1 = (const float*)d_in[2];
  const float* w2 = (const float*)d_in[3];
  const int*   ri = (const int*)d_in[4];
  float* out = (float*)d_out;
  char* ws = (char*)d_ws;

  u32*   amax = (u32*)(ws + OFF_AMAX);
  int*   pos  = (int*)(ws + OFF_POS);
  u16*   xq   = (u16*)(ws + OFF_XQ);
  u16*   w1q  = (u16*)(ws + OFF_W1Q);
  u16*   w2q  = (u16*)(ws + OFF_W2Q);
  u16*   xd   = (u16*)(ws + OFF_XD);
  u16*   actb = (u16*)(ws + OFF_ACTB);
  u16*   acts = (u16*)(ws + OFF_ACTS);
  float* h2   = (float*)(ws + OFF_H2);

  (void)hipMemsetAsync(ws + OFF_AMAX, 0, 256, stream);
  (void)hipMemsetAsync(ws + OFF_XD, 0, XD_B, stream);

  amax_f32_kernel<<<dim3(64, 1), 256, 0, stream>>>(hs, amax + 0, (size_t)TT * HH);
  amax_f32_kernel<<<dim3(64, 8), 256, 0, stream>>>(w1, amax + 1, (size_t)2 * IS * HH);
  amax_f32_kernel<<<dim3(64, 8), 256, 0, stream>>>(w2, amax + 9, (size_t)HH * IS);

  {
    size_t nb = (size_t)TT * HH / 16;
    quant_f32_kernel<<<(unsigned)((nb + 255) / 256), 256, 0, stream>>>(hs, xq, amax + 0, (size_t)TT * HH, nb);
  }
  {
    size_t nb = (size_t)EE * 2 * IS * HH / 16;
    quant_f32_kernel<<<(unsigned)((nb + 255) / 256), 256, 0, stream>>>(w1, w1q, amax + 1, (size_t)2 * IS * HH, nb);
  }
  {
    size_t nb = (size_t)EE * HH * IS / 16;
    quant_f32_kernel<<<(unsigned)((nb + 255) / 256), 256, 0, stream>>>(w2, w2q, amax + 9, (size_t)HH * IS, nb);
  }

  route_kernel<<<1, 32, 0, stream>>>(ri, pos);
  scatter_x_kernel<<<TT * KK, 256, 0, stream>>>(xq, ri, pos, xd);

  gemm1_act_kernel<<<dim3(IS / 64, CAP / 128, EE), 256, 0, stream>>>(xd, w1q, actb, amax, amax + 17);

  {
    size_t nb = (size_t)EE * CAP * IS / 16;
    quant_bf16_kernel<<<(unsigned)((nb + 255) / 256), 256, 0, stream>>>(actb, acts, amax + 17, (size_t)EE * CAP * IS, nb);
  }

  gemm2_kernel<<<dim3(HH / 64, CAP / 128, EE), 256, 0, stream>>>(acts, w2q, h2, amax);

  gather_out_kernel<<<(unsigned)(((size_t)TT * HH + 255) / 256), 256, 0, stream>>>(h2, ri, pos, rw, out);
}
